// ChemGeomFeatEncoder_51470888075538
// MI455X (gfx1250) — matrix operand order repaired and checked by reading, not compiled, not run on hardware
//
#include <hip/hip_runtime.h>
#include <hip/hip_bf16.h>

// ---------------------------------------------------------------------------
// Problem constants (match the reference)
// ---------------------------------------------------------------------------
#define H_   128
#define CF_  128
#define GF_  64
#define N_   500000
#define V_   100000
#define EPS_ 1e-5f

typedef __attribute__((ext_vector_type(16))) __bf16 v16bf;
typedef __attribute__((ext_vector_type(2)))  __bf16 v2bf;
typedef __attribute__((ext_vector_type(8)))  float  v8f;

union Bf16x16 { unsigned int d[8]; v16bf v; };
union Bf16Q   { uint4 q[2];        v16bf v; };

// f32 -> bf16 via native fptrunc (backend selects v_cvt*_bf16_f32 if present)
__device__ __forceinline__ unsigned short f2bf(float f) {
    __bf16 h = (__bf16)f;
    return __builtin_bit_cast(unsigned short, h);
}

// packed f32x2 -> bf16x2 (low = x, high = y); build-vector of fptruncs lets
// ISel pick v_cvt_pk_bf16_f32 when available.
__device__ __forceinline__ unsigned int pkbf(float x, float y) {
    v2bf r = { (__bf16)x, (__bf16)y };
    return __builtin_bit_cast(unsigned int, r);
}

// ---------------------------------------------------------------------------
// Fragment loaders
//
// A-layout (16x32 bf16, M x K): lanes 0-15 = rows, elems 0..7 -> K{k0..k0+7},
// elems 8..15 -> K{k0+16..k0+23}; lanes 16-31 same rows, K offset +8.
// B-layout (32x16 bf16, K x N): lane%16 = column n, same K order as A:
// elems 0..7 -> K{k0..k0+7}, elems 8..15 -> K{k0+16..k0+23}; lanes 16-31 K offset +8.
// Weights are pre-transposed AND pre-scaled by the folded-BN scale:
//   Wt[n][k] = bf16(W[k][n] * s[n]),  so  y = x@Wt^T + t[n].
// ---------------------------------------------------------------------------
__device__ __forceinline__ v16bf load_a_frag_f32(const float* A, int lda,
                                                 int m0, int k0, int lane) {
    int m  = lane & 15;
    int kh = (lane >> 4) * 8;
    const float* p = A + (size_t)(m0 + m) * lda + k0 + kh;
    float4 a = *(const float4*)(p);
    float4 b = *(const float4*)(p + 4);
    float4 c = *(const float4*)(p + 16);
    float4 d = *(const float4*)(p + 20);
    Bf16x16 u;
    u.d[0] = pkbf(a.x, a.y); u.d[1] = pkbf(a.z, a.w);
    u.d[2] = pkbf(b.x, b.y); u.d[3] = pkbf(b.z, b.w);
    u.d[4] = pkbf(c.x, c.y); u.d[5] = pkbf(c.z, c.w);
    u.d[6] = pkbf(d.x, d.y); u.d[7] = pkbf(d.z, d.w);
    return u.v;
}

__device__ __forceinline__ v16bf load_b_frag(const unsigned short* Wt, int ldk,
                                             int n0, int k0, int lane) {
    int n  = lane & 15;
    int ko = (lane >> 4) * 8;
    const uint4* p = (const uint4*)(Wt + (size_t)(n0 + n) * ldk + k0 + ko);
    Bf16Q u; u.q[0] = p[0]; u.q[1] = p[2];
    return u.v;
}

#define LP_ 136   // LDS row pitch in halves (272 B = 17*16 B: aligned + conflict-breaking)

__device__ __forceinline__ v16bf load_a_frag_lds(const unsigned short* buf,
                                                 int k0, int lane) {
    int m  = lane & 15;
    int kh = (lane >> 4) * 8;
    const uint4* p = (const uint4*)(buf + m * LP_ + k0 + kh);
    const uint4* q = (const uint4*)(buf + m * LP_ + k0 + 16 + kh);
    Bf16Q u; u.q[0] = *p; u.q[1] = *q;
    return u.v;
}

#define WMMA_BF16(A, B, C) \
    __builtin_amdgcn_wmma_f32_16x16x32_bf16(false, (A), false, (B), (short)0, (C), false, false)

__device__ __forceinline__ float silu_f(float y)  { return y / (1.f + __expf(-y)); }
__device__ __forceinline__ float sigm_f(float y)  { return 1.f / (1.f + __expf(-y)); }
__device__ __forceinline__ float softplus_f(float y) {
    return (y > 20.f) ? y : __logf(1.f + __expf(y));
}

__device__ __forceinline__ v8f bcast8(float t) {
    v8f a = {t, t, t, t, t, t, t, t};
    return a;
}

// ---------------------------------------------------------------------------
// Prep kernels
// ---------------------------------------------------------------------------
__global__ void zero_f32_kernel(float* p, long n) {
    long i = (long)blockIdx.x * blockDim.x + threadIdx.x;
    long s = (long)gridDim.x * blockDim.x;
    for (; i < n; i += s) p[i] = 0.f;
}

// Wt[n*K + k] = bf16(W[k*Nc + n] * s[n])   (transpose + fold BN scale)
__global__ void transpose_bf16_kernel(const float* __restrict__ W,
                                      const float* __restrict__ s,
                                      unsigned short* __restrict__ Wt,
                                      int K, int Nc) {
    int idx = blockIdx.x * blockDim.x + threadIdx.x;
    if (idx < K * Nc) {
        int n = idx / K, k = idx % K;
        Wt[idx] = f2bf(W[(size_t)k * Nc + n] * s[n]);
    }
}

// fold bias + batchnorm:  y = (x + b - m) * g*rsqrt(v+eps) + be  ->  (x@W)*s + t
__global__ void fold_bn_kernel(const float* __restrict__ b,
                               const float* __restrict__ g,
                               const float* __restrict__ be,
                               const float* __restrict__ m,
                               const float* __restrict__ v,
                               float* __restrict__ s, float* __restrict__ t, int n) {
    int i = blockIdx.x * blockDim.x + threadIdx.x;
    if (i < n) {
        float sc = g[i] * rsqrtf(v[i] + EPS_);
        s[i] = sc;
        t[i] = (b[i] - m[i]) * sc + be[i];
    }
}

// ---------------------------------------------------------------------------
// chem MLP: (N,128) -> 128 -> silu -> 256 -> sigmoid(f)*softplus(c)
//           writes h_chem (N,128) and atomically accumulates segment sum.
// ---------------------------------------------------------------------------
__global__ __launch_bounds__(256)
void chem_kernel(const float* __restrict__ A, const int* __restrict__ vids,
                 const unsigned short* __restrict__ W1t,
                 const unsigned short* __restrict__ W2t,
                 const float* __restrict__ t1, const float* __restrict__ t2,
                 float* __restrict__ hchem, float* __restrict__ hcg) {
    __shared__ unsigned short hb[8 * 16 * LP_];
    int wave = threadIdx.x >> 5, lane = threadIdx.x & 31;
    int m0 = (blockIdx.x * 8 + wave) * 16;
    if (m0 >= N_) return;
    unsigned short* buf = hb + wave * 16 * LP_;

    // A fragments for full K=128 row-tile, kept in registers
    v16bf a1[4];
    #pragma unroll
    for (int k = 0; k < 4; ++k) a1[k] = load_a_frag_f32(A, CF_, m0, k * 32, lane);

    // ---- GEMM1 (+folded bias/BN) + SiLU -> LDS (bf16) ----
    #pragma unroll
    for (int j = 0; j < 8; ++j) {
        int n = j * 16 + (lane & 15);
        v8f acc = bcast8(t1[n]);
        #pragma unroll
        for (int k = 0; k < 4; ++k)
            acc = WMMA_BF16(a1[k], load_b_frag(W1t, CF_, j * 16, k * 32, lane), acc);
        #pragma unroll
        for (int r = 0; r < 8; ++r) {
            float y = silu_f(acc[r]);
            int row = r + ((lane >> 4) << 3);
            buf[row * LP_ + n] = f2bf(y);
        }
    }
    asm volatile("s_wait_dscnt 0" ::: "memory");

    // ---- GEMM2 (256 outputs): filter tile j pairs with core tile j+8 ----
    v16bf a2[4];
    #pragma unroll
    for (int k = 0; k < 4; ++k) a2[k] = load_a_frag_lds(buf, k * 32, lane);

    int rbase = m0 + ((lane >> 4) << 3);
    int vrow[8];
    #pragma unroll
    for (int r = 0; r < 8; ++r) vrow[r] = vids[rbase + r];

    #pragma unroll
    for (int j = 0; j < 8; ++j) {
        int col = j * 16 + (lane & 15);
        v8f accF = bcast8(t2[col]);
        v8f accC = bcast8(t2[128 + col]);
        #pragma unroll
        for (int k = 0; k < 4; ++k) {
            accF = WMMA_BF16(a2[k], load_b_frag(W2t, H_, j * 16,       k * 32, lane), accF);
            accC = WMMA_BF16(a2[k], load_b_frag(W2t, H_, 128 + j * 16, k * 32, lane), accC);
        }
        #pragma unroll
        for (int r = 0; r < 8; ++r) {
            float h = sigm_f(accF[r]) * softplus_f(accC[r]);
            int row = rbase + r;
            hchem[(size_t)row * H_ + col] = h;
            atomicAdd(&hcg[(size_t)vrow[r] * H_ + col], h);
        }
    }
}

// ---------------------------------------------------------------------------
// geom MLP: (V,64) -> 128 -> silu -> 128, result (f32) to workspace
// ---------------------------------------------------------------------------
__global__ __launch_bounds__(256)
void geom_kernel(const float* __restrict__ A,
                 const unsigned short* __restrict__ W1t,
                 const unsigned short* __restrict__ W2t,
                 const float* __restrict__ t1, const float* __restrict__ t2,
                 float* __restrict__ out) {
    __shared__ unsigned short hb[8 * 16 * LP_];
    int wave = threadIdx.x >> 5, lane = threadIdx.x & 31;
    int m0 = (blockIdx.x * 8 + wave) * 16;
    if (m0 >= V_) return;
    unsigned short* buf = hb + wave * 16 * LP_;

    v16bf a1[2];
    #pragma unroll
    for (int k = 0; k < 2; ++k) a1[k] = load_a_frag_f32(A, GF_, m0, k * 32, lane);

    #pragma unroll
    for (int j = 0; j < 8; ++j) {
        int n = j * 16 + (lane & 15);
        v8f acc = bcast8(t1[n]);
        #pragma unroll
        for (int k = 0; k < 2; ++k)
            acc = WMMA_BF16(a1[k], load_b_frag(W1t, GF_, j * 16, k * 32, lane), acc);
        #pragma unroll
        for (int r = 0; r < 8; ++r) {
            float y = silu_f(acc[r]);
            int row = r + ((lane >> 4) << 3);
            buf[row * LP_ + n] = f2bf(y);
        }
    }
    asm volatile("s_wait_dscnt 0" ::: "memory");

    v16bf a2[4];
    #pragma unroll
    for (int k = 0; k < 4; ++k) a2[k] = load_a_frag_lds(buf, k * 32, lane);

    int rbase = m0 + ((lane >> 4) << 3);
    #pragma unroll
    for (int j = 0; j < 8; ++j) {
        int col = j * 16 + (lane & 15);
        v8f acc = bcast8(t2[col]);
        #pragma unroll
        for (int k = 0; k < 4; ++k)
            acc = WMMA_BF16(a2[k], load_b_frag(W2t, H_, j * 16, k * 32, lane), acc);
        #pragma unroll
        for (int r = 0; r < 8; ++r)
            out[(size_t)(rbase + r) * H_ + col] = acc[r];
    }
}

// ---------------------------------------------------------------------------
// final MLP: concat(hcg, hgeom) (V,256) -> 128 -> silu -> 128 -> d_out[0:V*128]
// ---------------------------------------------------------------------------
__global__ __launch_bounds__(256)
void final_kernel(const float* __restrict__ Acg, const float* __restrict__ Agm,
                  const unsigned short* __restrict__ W1t,
                  const unsigned short* __restrict__ W2t,
                  const float* __restrict__ t1, const float* __restrict__ t2,
                  float* __restrict__ out) {
    __shared__ unsigned short hb[8 * 16 * LP_];
    int wave = threadIdx.x >> 5, lane = threadIdx.x & 31;
    int m0 = (blockIdx.x * 8 + wave) * 16;
    if (m0 >= V_) return;
    unsigned short* buf = hb + wave * 16 * LP_;

    v16bf a1[8];
    #pragma unroll
    for (int k = 0; k < 4; ++k) a1[k]     = load_a_frag_f32(Acg, H_, m0, k * 32, lane);
    #pragma unroll
    for (int k = 0; k < 4; ++k) a1[k + 4] = load_a_frag_f32(Agm, H_, m0, k * 32, lane);

    #pragma unroll
    for (int j = 0; j < 8; ++j) {
        int n = j * 16 + (lane & 15);
        v8f acc = bcast8(t1[n]);
        #pragma unroll
        for (int k = 0; k < 8; ++k)
            acc = WMMA_BF16(a1[k], load_b_frag(W1t, 2 * H_, j * 16, k * 32, lane), acc);
        #pragma unroll
        for (int r = 0; r < 8; ++r) {
            float y = silu_f(acc[r]);
            int row = r + ((lane >> 4) << 3);
            buf[row * LP_ + n] = f2bf(y);
        }
    }
    asm volatile("s_wait_dscnt 0" ::: "memory");

    v16bf a2[4];
    #pragma unroll
    for (int k = 0; k < 4; ++k) a2[k] = load_a_frag_lds(buf, k * 32, lane);

    int rbase = m0 + ((lane >> 4) << 3);
    #pragma unroll
    for (int j = 0; j < 8; ++j) {
        int col = j * 16 + (lane & 15);
        v8f acc = bcast8(t2[col]);
        #pragma unroll
        for (int k = 0; k < 4; ++k)
            acc = WMMA_BF16(a2[k], load_b_frag(W2t, H_, j * 16, k * 32, lane), acc);
        #pragma unroll
        for (int r = 0; r < 8; ++r)
            out[(size_t)(rbase + r) * H_ + col] = acc[r];
    }
}

// ---------------------------------------------------------------------------
// Host launcher
// ---------------------------------------------------------------------------
extern "C" void kernel_launch(void* const* d_in, const int* in_sizes, int n_in,
                              void* d_out, int out_size, void* d_ws, size_t ws_size,
                              hipStream_t stream) {
    (void)in_sizes; (void)n_in; (void)out_size; (void)ws_size;

    const float* chem = (const float*)d_in[0];
    const float* geom = (const float*)d_in[1];
    const int*   vids = (const int*)d_in[2];
    const float* cW1 = (const float*)d_in[3];  const float* cb1 = (const float*)d_in[4];
    const float* cg1 = (const float*)d_in[5];  const float* cbe1 = (const float*)d_in[6];
    const float* cm1 = (const float*)d_in[7];  const float* cv1 = (const float*)d_in[8];
    const float* cW2 = (const float*)d_in[9];  const float* cb2 = (const float*)d_in[10];
    const float* cg2 = (const float*)d_in[11]; const float* cbe2 = (const float*)d_in[12];
    const float* cm2 = (const float*)d_in[13]; const float* cv2 = (const float*)d_in[14];
    const float* gW1 = (const float*)d_in[15]; const float* gb1 = (const float*)d_in[16];
    const float* gg1 = (const float*)d_in[17]; const float* gbe1 = (const float*)d_in[18];
    const float* gm1 = (const float*)d_in[19]; const float* gv1 = (const float*)d_in[20];
    const float* gW2 = (const float*)d_in[21]; const float* gb2 = (const float*)d_in[22];
    const float* gg2 = (const float*)d_in[23]; const float* gbe2 = (const float*)d_in[24];
    const float* gm2 = (const float*)d_in[25]; const float* gv2 = (const float*)d_in[26];
    const float* fW1 = (const float*)d_in[27]; const float* fb1 = (const float*)d_in[28];
    const float* fg1 = (const float*)d_in[29]; const float* fbe1 = (const float*)d_in[30];
    const float* fm1 = (const float*)d_in[31]; const float* fv1 = (const float*)d_in[32];
    const float* fW2 = (const float*)d_in[33]; const float* fb2 = (const float*)d_in[34];
    const float* fg2 = (const float*)d_in[35]; const float* fbe2 = (const float*)d_in[36];
    const float* fm2 = (const float*)d_in[37]; const float* fv2 = (const float*)d_in[38];

    float* out = (float*)d_out;             // [0, V*H) = h_geom ; [V*H, V*H+N*H) = h_chem

    // workspace layout
    float* hcg   = (float*)d_ws;                        // V*H f32 (segment sum)
    float* hgeom = hcg + (size_t)V_ * H_;               // V*H f32
    unsigned short* cW1t = (unsigned short*)(hgeom + (size_t)V_ * H_);
    unsigned short* cW2t = cW1t + H_ * CF_;             // 256 x 128
    unsigned short* gW1t = cW2t + 2 * H_ * H_;          // 128 x 64
    unsigned short* gW2t = gW1t + H_ * GF_;             // 128 x 128
    unsigned short* fW1t = gW2t + H_ * H_;              // 128 x 256
    unsigned short* fW2t = fW1t + 2 * H_ * H_;          // 128 x 128
    float* sb = (float*)(fW2t + H_ * H_);
    float *c1s = sb,        *c1t = sb + 128;
    float *c2s = sb + 256,  *c2t = sb + 512;
    float *g1s = sb + 768,  *g1t = sb + 896;
    float *g2s = sb + 1024, *g2t = sb + 1152;
    float *f1s = sb + 1280, *f1t = sb + 1408;
    float *f2s = sb + 1536, *f2t = sb + 1664;

    // ---- prep: fold BN first (scales feed the weight transposes) ----
    zero_f32_kernel<<<2048, 256, 0, stream>>>(hcg, (long)V_ * H_);

    fold_bn_kernel<<<1, 128, 0, stream>>>(cb1, cg1, cbe1, cm1, cv1, c1s, c1t, H_);
    fold_bn_kernel<<<2, 128, 0, stream>>>(cb2, cg2, cbe2, cm2, cv2, c2s, c2t, 2 * H_);
    fold_bn_kernel<<<1, 128, 0, stream>>>(gb1, gg1, gbe1, gm1, gv1, g1s, g1t, H_);
    fold_bn_kernel<<<1, 128, 0, stream>>>(gb2, gg2, gbe2, gm2, gv2, g2s, g2t, H_);
    fold_bn_kernel<<<1, 128, 0, stream>>>(fb1, fg1, fbe1, fm1, fv1, f1s, f1t, H_);
    fold_bn_kernel<<<1, 128, 0, stream>>>(fb2, fg2, fbe2, fm2, fv2, f2s, f2t, H_);

    transpose_bf16_kernel<<<(CF_ * H_ + 255) / 256, 256, 0, stream>>>(cW1, c1s, cW1t, CF_, H_);
    transpose_bf16_kernel<<<(H_ * 2 * H_ + 255) / 256, 256, 0, stream>>>(cW2, c2s, cW2t, H_, 2 * H_);
    transpose_bf16_kernel<<<(GF_ * H_ + 255) / 256, 256, 0, stream>>>(gW1, g1s, gW1t, GF_, H_);
    transpose_bf16_kernel<<<(H_ * H_ + 255) / 256, 256, 0, stream>>>(gW2, g2s, gW2t, H_, H_);
    transpose_bf16_kernel<<<(2 * H_ * H_ + 255) / 256, 256, 0, stream>>>(fW1, f1s, fW1t, 2 * H_, H_);
    transpose_bf16_kernel<<<(H_ * H_ + 255) / 256, 256, 0, stream>>>(fW2, f2s, fW2t, H_, H_);

    // ---- compute ----
    int chem_blocks = (N_ / 16 + 7) / 8;   // 3907
    int vtx_blocks  = (V_ / 16 + 7) / 8;   // 782

    chem_kernel<<<chem_blocks, 256, 0, stream>>>(chem, vids, cW1t, cW2t,
                                                 c1t, c2t,
                                                 out + (size_t)V_ * H_, hcg);
    geom_kernel<<<vtx_blocks, 256, 0, stream>>>(geom, gW1t, gW2t,
                                                g1t, g2t, hgeom);
    final_kernel<<<vtx_blocks, 256, 0, stream>>>(hcg, hgeom, fW1t, fW2t,
                                                 f1t, f2t, out);
}
